// ScaledDotProductAttention_32959579030342
// MI455X (gfx1250) — matrix operand order repaired and checked by reading, not compiled, not run on hardware
//
#include <hip/hip_runtime.h>
#include <stdint.h>

typedef __attribute__((ext_vector_type(16))) _Float16 v16h;
typedef __attribute__((ext_vector_type(8)))  float    v8f;

#define BATCH 16
#define SEQ   2048
#define DIM   128
#define BR    128              // query rows per block (8 waves x 16 rows)
#define BC    64               // kv tile depth per iteration
#define NTILE (SEQ / BC)       // 32
#define NWAVE 8

// ---------------- pre-pass: fp32 -> f16, V transposed per kv tile ----------------
// Kh  layout: [b][s][d]                       (row-major f16)
// Vth layout: [b][tile][d][key]  key in 0..63 (per-tile transposed f16)
__global__ __launch_bounds__(256)
void cvt_kv_kernel(const float* __restrict__ K, const float* __restrict__ V,
                   _Float16* __restrict__ Kh, _Float16* __restrict__ Vth) {
  const int idx4 = blockIdx.x * 256 + threadIdx.x;   // float4 index
  const int lin  = idx4 * 4;

  const float4 k4 = ((const float4*)K)[idx4];
  Kh[lin + 0] = (_Float16)k4.x;
  Kh[lin + 1] = (_Float16)k4.y;
  Kh[lin + 2] = (_Float16)k4.z;
  Kh[lin + 3] = (_Float16)k4.w;

  const float4 v4 = ((const float4*)V)[idx4];
  const int b    = lin / (SEQ * DIM);
  const int rem  = lin % (SEQ * DIM);
  const int s    = rem / DIM;
  const int d0   = rem % DIM;
  const int tile = s / BC;
  const int key  = s % BC;
  _Float16* dst = Vth + (((size_t)b * NTILE + tile) * DIM + d0) * BC + key;
  dst[0 * BC] = (_Float16)v4.x;
  dst[1 * BC] = (_Float16)v4.y;
  dst[2 * BC] = (_Float16)v4.z;
  dst[3 * BC] = (_Float16)v4.w;
}

// ---------------- main flash-attention kernel ----------------
__device__ __forceinline__ float rowReduceMax(float x) {
  x = fmaxf(x, __shfl_xor(x, 1, 32));
  x = fmaxf(x, __shfl_xor(x, 2, 32));
  x = fmaxf(x, __shfl_xor(x, 4, 32));
  x = fmaxf(x, __shfl_xor(x, 8, 32));
  return x;
}
__device__ __forceinline__ float rowReduceSum(float x) {
  x += __shfl_xor(x, 1, 32);
  x += __shfl_xor(x, 2, 32);
  x += __shfl_xor(x, 4, 32);
  x += __shfl_xor(x, 8, 32);
  return x;
}

__global__ __launch_bounds__(256)
void fa_wmma_kernel(const float* __restrict__ Q, const _Float16* __restrict__ Kh,
                    const _Float16* __restrict__ Vth, float* __restrict__ O) {
  __shared__ _Float16 Kbuf[2][BC * DIM];         // double-buffered K tile [key][d]
  __shared__ _Float16 Vbuf[2][DIM * BC];         // double-buffered V^T tile [d][key]
  __shared__ _Float16 Plds[NWAVE * 16 * BC];     // per-wave P staging [row][col]

  const int tid    = threadIdx.x;
  const int wave   = tid >> 5;
  const int lane   = tid & 31;
  const int laneLo = lane & 15;
  const int laneHi = lane >> 4;   // 0 or 1 (half-wave)

  const int b  = blockIdx.x / (SEQ / BR);
  const int q0 = (blockIdx.x % (SEQ / BR)) * BR;
  const int qrow = q0 + wave * 16 + laneLo;

  const float qscale = 0.088388347648318447f;    // 1/sqrt(128)

  // ---- Load Q fragments in WMMA A layout (fp32 -> f16, pre-scaled) ----
  v16h qfrag[4];
  const float* qg = Q + ((size_t)b * SEQ + qrow) * DIM;
#pragma unroll
  for (int t = 0; t < 4; ++t) {
#pragma unroll
    for (int i = 0; i < 8; ++i) {
      const int kb = ((i < 4) ? 2 * i : 16 + 2 * (i - 4)) + 8 * laneHi + 32 * t;
      qfrag[t][2 * i]     = (_Float16)(qg[kb]     * qscale);
      qfrag[t][2 * i + 1] = (_Float16)(qg[kb + 1] * qscale);
    }
  }

  v8f acc[8];
#pragma unroll
  for (int i = 0; i < 8; ++i) acc[i] = (v8f){0.f,0.f,0.f,0.f,0.f,0.f,0.f,0.f};
  float mrow[8], lrow[8];
#pragma unroll
  for (int i = 0; i < 8; ++i) { mrow[i] = -1e30f; lrow[i] = 0.f; }

  _Float16* Pmine = Plds + wave * (16 * BC);

  const uint64_t kbase = (uint64_t)(uintptr_t)(Kh  + (size_t)b * SEQ * DIM);
  const uint64_t vbase = (uint64_t)(uintptr_t)(Vth + (size_t)b * NTILE * DIM * BC);

  // Issue one tile's async DMA: 2 x 16KB contiguous -> LDS, 8 b128 ops/lane total
  auto issueTile = [&](int tile, int bufIdx) {
    const uint64_t kt = kbase + (uint64_t)tile * (BC * DIM * 2);
    const uint64_t vt = vbase + (uint64_t)tile * (DIM * BC * 2);
    const uint32_t ldsK = (uint32_t)(uintptr_t)(&Kbuf[bufIdx][0]);
    const uint32_t ldsV = (uint32_t)(uintptr_t)(&Vbuf[bufIdx][0]);
#pragma unroll
    for (int c = 0; c < 4; ++c) {
      const uint32_t off = (uint32_t)(tid + c * 256) * 16u;
      asm volatile("global_load_async_to_lds_b128 %0, %1, %2"
                   :: "v"(ldsK + off), "v"(off), "s"(kt) : "memory");
    }
#pragma unroll
    for (int c = 0; c < 4; ++c) {
      const uint32_t off = (uint32_t)(tid + c * 256) * 16u;
      asm volatile("global_load_async_to_lds_b128 %0, %1, %2"
                   :: "v"(ldsV + off), "v"(off), "s"(vt) : "memory");
    }
  };

  issueTile(0, 0);

  for (int t = 0; t < NTILE; ++t) {
    if (t + 1 < NTILE) {
      issueTile(t + 1, (t + 1) & 1);
      // this tile's 8 async ops done; next tile's 8 still in flight
      asm volatile("s_wait_asynccnt 0x8" ::: "memory");
    } else {
      asm volatile("s_wait_asynccnt 0x0" ::: "memory");
    }
    __syncthreads();

    const _Float16* Kt = &Kbuf[t & 1][0];
    const _Float16* Vt = &Vbuf[t & 1][0];

    // ---- S = Q @ K^T : 16x64 scores (4 N-subtiles x 4 K-steps) ----
    v8f sc[4];
#pragma unroll
    for (int i = 0; i < 4; ++i) sc[i] = (v8f){0.f,0.f,0.f,0.f,0.f,0.f,0.f,0.f};
#pragma unroll
    for (int nt = 0; nt < 4; ++nt) {
      const int key = nt * 16 + laneLo;
#pragma unroll
      for (int kt = 0; kt < 4; ++kt) {
        const _Float16* src = Kt + key * DIM + 8 * laneHi + 32 * kt;
        v16h bfrag;
#pragma unroll
        for (int e = 0; e < 16; ++e) bfrag[e] = src[(e < 8) ? e : e + 8];
        sc[nt] = __builtin_amdgcn_wmma_f32_16x16x32_f16(
            false, qfrag[kt], false, bfrag, (short)0, sc[nt], false, false);
      }
    }

    // ---- Online softmax (C layout: row = v + 8*laneHi, col = laneLo + 16*nt) ----
#pragma unroll
    for (int v = 0; v < 8; ++v) {
      float sm = fmaxf(fmaxf(sc[0][v], sc[1][v]), fmaxf(sc[2][v], sc[3][v]));
      sm = rowReduceMax(sm);
      const float newm  = fmaxf(mrow[v], sm);
      const float scale = __expf(mrow[v] - newm);
      float p[4], psum = 0.f;
#pragma unroll
      for (int j = 0; j < 4; ++j) { p[j] = __expf(sc[j][v] - newm); psum += p[j]; }
      lrow[v] = lrow[v] * scale + rowReduceSum(psum);
      mrow[v] = newm;
#pragma unroll
      for (int nt = 0; nt < 8; ++nt) acc[nt][v] *= scale;
      const int row = v + 8 * laneHi;
#pragma unroll
      for (int j = 0; j < 4; ++j)
        Pmine[row * BC + j * 16 + laneLo] = (_Float16)p[j];
    }

    // wave-local LDS RAW on P staging
    asm volatile("s_wait_dscnt 0x0" ::: "memory");

    v16h pfrag[2];
#pragma unroll
    for (int kt = 0; kt < 2; ++kt) {
#pragma unroll
      for (int i = 0; i < 8; ++i) {
        const int kb = ((i < 4) ? 2 * i : 16 + 2 * (i - 4)) + 8 * laneHi + 32 * kt;
        pfrag[kt][2 * i]     = Pmine[laneLo * BC + kb];
        pfrag[kt][2 * i + 1] = Pmine[laneLo * BC + kb + 1];
      }
    }

    // ---- O += P @ V : 8 feature subtiles x 2 K-steps ----
#pragma unroll
    for (int nt = 0; nt < 8; ++nt) {
      const int feat = nt * 16 + laneLo;
#pragma unroll
      for (int kt = 0; kt < 2; ++kt) {
        const _Float16* src = Vt + feat * BC + 8 * laneHi + 32 * kt;
        v16h vb;
#pragma unroll
        for (int e = 0; e < 16; ++e) vb[e] = src[(e < 8) ? e : e + 8];
        acc[nt] = __builtin_amdgcn_wmma_f32_16x16x32_f16(
            false, pfrag[kt], false, vb, (short)0, acc[nt], false, false);
      }
    }
    __syncthreads();   // everyone done with buf[t&1] before tile t+2 overwrites it
  }

  // ---- Epilogue: normalize by row sum, store fp32 ----
  float* og = O + ((size_t)b * SEQ + q0 + wave * 16) * DIM;
#pragma unroll
  for (int v = 0; v < 8; ++v) {
    const float inv = 1.0f / lrow[v];
    const int row = v + 8 * laneHi;
#pragma unroll
    for (int nt = 0; nt < 8; ++nt) {
      og[row * DIM + nt * 16 + laneLo] = acc[nt][v] * inv;
    }
  }
}

extern "C" void kernel_launch(void* const* d_in, const int* in_sizes, int n_in,
                              void* d_out, int out_size, void* d_ws, size_t ws_size,
                              hipStream_t stream) {
  const float* Q = (const float*)d_in[0];
  const float* K = (const float*)d_in[1];
  const float* V = (const float*)d_in[2];
  float* O = (float*)d_out;
  (void)in_sizes; (void)n_in; (void)out_size; (void)ws_size;

  // workspace partition: Kh (8 MB f16) | Vth (8 MB f16)
  _Float16* Kh  = (_Float16*)d_ws;
  _Float16* Vth = Kh + (size_t)BATCH * SEQ * DIM;

  const int cvt_blocks = (BATCH * SEQ * DIM / 4) / 256;  // 4096
  cvt_kv_kernel<<<cvt_blocks, 256, 0, stream>>>(K, V, Kh, Vth);

  dim3 grid(BATCH * (SEQ / BR));   // 256 blocks
  fa_wmma_kernel<<<grid, 256, 0, stream>>>(Q, Kh, Vth, O);
}
